// ImageCaptionModel_28003186770564
// MI455X (gfx1250) — compile-verified, matrix operand order checked by reading, not run on hardware
//
#include <hip/hip_runtime.h>
#include <hip/hip_bf16.h>

// ---------------------------------------------------------------------------
// Problem constants (from reference): B=64 S=32 V=10000 E=H=512 F=2048 L=2
// ---------------------------------------------------------------------------
#define BB 64
#define SS 32
#define VV 10000
#define EE 512
#define HH 512
#define FF 2048
#define DD 1024   // E + H
#define LL 2

typedef __bf16 bf16_t;
typedef __attribute__((ext_vector_type(16))) __bf16 v16bf;
typedef __attribute__((ext_vector_type(8)))  __bf16 v8bf;
typedef __attribute__((ext_vector_type(8)))  float  v8f;

__device__ __forceinline__ float sigmoidf_(float x) {
    return 1.0f / (1.0f + __expf(-x));
}

// ---------------------------------------------------------------------------
// WMMA helpers (CDNA5 wave32, V_WMMA_F32_16X16X32_BF16).
// A fragment: 16x32 (MxK) bf16; lane<16 holds row M=lane, K={kb..kb+7, kb+16..kb+23};
// lane>=16 holds row M=lane-16, K={kb+8..kb+15, kb+24..kb+31}.  Two b128 loads/lane.
// B fragment (from pre-transposed W^T [N,K]) uses the identical pattern with
// rows replaced by output columns.
// ---------------------------------------------------------------------------
__device__ __forceinline__ v16bf load_tile16x32(const bf16_t* __restrict__ base,
                                                long rowStride, int r0, int kbase,
                                                int lane) {
    const int li = lane & 15;
    const int ko = (lane >> 4) << 3;          // 0 or 8
    const bf16_t* p = base + (long)(r0 + li) * rowStride + (kbase + ko);
    v8bf lo = *(const v8bf*)(p);
    v8bf hi = *(const v8bf*)(p + 16);
    v16bf a;
#pragma unroll
    for (int i = 0; i < 8; ++i) { a[i] = lo[i]; a[i + 8] = hi[i]; }
    return a;
}

__device__ __forceinline__ v8f wmma_bf16(v16bf a, v16bf b, v8f c) {
    // (neg_a, A, neg_b, B, c_mod, C, reuse_a, reuse_b)
    return __builtin_amdgcn_wmma_f32_16x16x32_bf16(false, a, false, b,
                                                   (short)0, c, false, false);
}

// ---------------------------------------------------------------------------
// f32 -> bf16 elementwise convert
// ---------------------------------------------------------------------------
__global__ void k_f32_to_bf16(const float* __restrict__ src,
                              bf16_t* __restrict__ dst, int n) {
    int i = blockIdx.x * blockDim.x + threadIdx.x;
    if (i < n) dst[i] = (bf16_t)src[i];
}

// ---------------------------------------------------------------------------
// LDS-tiled transpose + convert: src f32 [K,N] -> dst bf16 [N,K]
// block: 32x8, grid: (ceil(N/32), ceil(K/32))
// ---------------------------------------------------------------------------
__global__ void k_transpose_f32_bf16(const float* __restrict__ src,
                                     bf16_t* __restrict__ dst, int K, int N) {
    __shared__ bf16_t tile[32][33];
    const int nb = blockIdx.x * 32;
    const int kb = blockIdx.y * 32;
    const int tx = threadIdx.x, ty = threadIdx.y;
#pragma unroll
    for (int i = 0; i < 32; i += 8) {
        int k = kb + ty + i, n = nb + tx;
        if (k < K && n < N) tile[ty + i][tx] = (bf16_t)src[(long)k * N + n];
    }
    __syncthreads();
#pragma unroll
    for (int i = 0; i < 32; i += 8) {
        int n = nb + ty + i, k = kb + tx;
        if (n < N && k < K) dst[(long)n * K + k] = tile[tx][ty + i];
    }
}

// ---------------------------------------------------------------------------
// Embedding gather + convert: seq[b,s,:] = bf16(emb[tokens[b,s],:])
// ---------------------------------------------------------------------------
__global__ void k_embed_gather(const int* __restrict__ tokens,
                               const float* __restrict__ emb,
                               bf16_t* __restrict__ seq) {
    const int bs = blockIdx.x;                 // 0..B*S-1
    const int tok = tokens[bs];
    const float* src = emb + (long)tok * EE;
    bf16_t* dst = seq + (long)bs * EE;
    for (int e = threadIdx.x; e < EE; e += blockDim.x) dst[e] = (bf16_t)src[e];
}

// ---------------------------------------------------------------------------
// h0 = cnn_bf16 [64x2048] @ w_inT^T + b_in  -> h0_f32, h0_bf16  (M=64,N=512,K=2048)
// 128 tiles of 16x16; 8 waves/block, 16 blocks.
// ---------------------------------------------------------------------------
__global__ void k_wmma_h0(const bf16_t* __restrict__ A,
                          const bf16_t* __restrict__ BT,  // [512, 2048]
                          const float* __restrict__ bias,
                          float* __restrict__ h0f, bf16_t* __restrict__ h0b) {
    const int lane = threadIdx.x & 31;
    const int wave = threadIdx.x >> 5;
    const int tile = blockIdx.x * 8 + wave;    // 0..127
    const int m0 = (tile >> 5) << 4;
    const int n0 = (tile & 31) << 4;

    v8f c = {};
    for (int k0 = 0; k0 < FF; k0 += 32) {
        v16bf a = load_tile16x32(A, FF, m0, k0, lane);
        v16bf b = load_tile16x32(BT, FF, n0, k0, lane);
        c = wmma_bf16(a, b, c);
    }
    const int li = lane & 15;
    const int ro = (lane >> 4) << 3;
    const int col = n0 + li;
    const float bv = bias[col];
#pragma unroll
    for (int i = 0; i < 8; ++i) {
        int row = m0 + ro + i;
        float v = c[i] + bv;
        h0f[(long)row * HH + col] = v;
        h0b[(long)row * HH + col] = (bf16_t)v;
    }
}

// ---------------------------------------------------------------------------
// GRU step part 1: r and u gates.
//   pre = concat(xt, h) @ W{r,u} + b{r,u}   (M=64, N=512, K=1024)
//   r -> rh_bf16 = bf16(sigmoid(pre_r) * h_f32)     (feeds candidate GEMM)
//   u -> u_f32   = sigmoid(pre_u)
// 256 wave-jobs (2 gates x 128 tiles); 8 waves/block, 32 blocks.
// ---------------------------------------------------------------------------
__global__ void k_gru_gates_ru(const bf16_t* __restrict__ xt, long xtStride,
                               const bf16_t* __restrict__ hbf,
                               const bf16_t* __restrict__ wrT,   // [512,1024]
                               const bf16_t* __restrict__ wuT,   // [512,1024]
                               const float* __restrict__ br,
                               const float* __restrict__ bu,
                               const float* __restrict__ hf,
                               float* __restrict__ uf,
                               bf16_t* __restrict__ rh) {
    const int lane = threadIdx.x & 31;
    const int wave = threadIdx.x >> 5;
    const int job  = blockIdx.x * 8 + wave;    // 0..255
    const int gate = job >> 7;                 // 0=r, 1=u
    const int tile = job & 127;
    const int m0 = (tile >> 5) << 4;
    const int n0 = (tile & 31) << 4;

    const bf16_t* BT = gate ? wuT : wrT;
    v8f c = {};
    for (int k0 = 0; k0 < DD; k0 += 32) {
        const bf16_t* ab;
        long astr;
        int kk;
        if (k0 < EE) { ab = xt;  astr = xtStride; kk = k0; }
        else         { ab = hbf; astr = HH;       kk = k0 - EE; }
        v16bf a = load_tile16x32(ab, astr, m0, kk, lane);
        v16bf b = load_tile16x32(BT, DD, n0, k0, lane);
        c = wmma_bf16(a, b, c);
    }
    const int li = lane & 15;
    const int ro = (lane >> 4) << 3;
    const int col = n0 + li;
    const float bv = gate ? bu[col] : br[col];
#pragma unroll
    for (int i = 0; i < 8; ++i) {
        int row = m0 + ro + i;
        long idx = (long)row * HH + col;
        float g = sigmoidf_(c[i] + bv);
        if (gate) uf[idx] = g;
        else      rh[idx] = (bf16_t)(g * hf[idx]);
    }
}

// ---------------------------------------------------------------------------
// GRU step part 2: candidate + state update.
//   h~ = tanh(concat(xt, r*h) @ Wh + bh)
//   h  = h*u + (1-u)*h~   -> h_f32, h_bf16, seq_out[:,t,:]
// 128 tiles; 8 waves/block, 16 blocks.
// ---------------------------------------------------------------------------
__global__ void k_gru_htilde_update(const bf16_t* __restrict__ xt, long xtStride,
                                    const bf16_t* __restrict__ rh,
                                    const bf16_t* __restrict__ whT,  // [512,1024]
                                    const float* __restrict__ bh,
                                    const float* __restrict__ uf,
                                    float* __restrict__ hf,
                                    bf16_t* __restrict__ hbf,
                                    bf16_t* __restrict__ seq_out_t, long seqStride) {
    const int lane = threadIdx.x & 31;
    const int wave = threadIdx.x >> 5;
    const int tile = blockIdx.x * 8 + wave;    // 0..127
    const int m0 = (tile >> 5) << 4;
    const int n0 = (tile & 31) << 4;

    v8f c = {};
    for (int k0 = 0; k0 < DD; k0 += 32) {
        const bf16_t* ab;
        long astr;
        int kk;
        if (k0 < EE) { ab = xt; astr = xtStride; kk = k0; }
        else         { ab = rh; astr = HH;       kk = k0 - EE; }
        v16bf a = load_tile16x32(ab, astr, m0, kk, lane);
        v16bf b = load_tile16x32(whT, DD, n0, k0, lane);
        c = wmma_bf16(a, b, c);
    }
    const int li = lane & 15;
    const int ro = (lane >> 4) << 3;
    const int col = n0 + li;
    const float bv = bh[col];
#pragma unroll
    for (int i = 0; i < 8; ++i) {
        int row = m0 + ro + i;
        long idx = (long)row * HH + col;
        float ht = tanhf(c[i] + bv);
        float u  = uf[idx];
        float hn = hf[idx] * u + (1.0f - u) * ht;
        hf[idx]  = hn;
        bf16_t hb = (bf16_t)hn;
        hbf[idx] = hb;
        seq_out_t[(long)row * seqStride + col] = hb;
    }
}

// ---------------------------------------------------------------------------
// Logits: seq [2048x512] bf16 @ w_out + b_out -> f32 [2048x10000]
// Per-wave 16x64 tile (A fragment reused across 4 WMMAs).
// N tiles = 625 -> 157 groups of 4; jobs = 128*157 = 20096; 8 waves/block.
// ---------------------------------------------------------------------------
__global__ void k_wmma_logits(const bf16_t* __restrict__ A,    // [2048,512]
                              const bf16_t* __restrict__ BT,   // [10000,512]
                              const float* __restrict__ bias,
                              float* __restrict__ out) {
    const int lane = threadIdx.x & 31;
    const int wave = threadIdx.x >> 5;
    const int job  = blockIdx.x * 8 + wave;    // 0..20095
    const int mt = job / 157;
    const int ng = job % 157;
    const int m0 = mt << 4;

    v8f c[4] = {};
    const int li = lane & 15;
    for (int k0 = 0; k0 < HH; k0 += 32) {
        v16bf a = load_tile16x32(A, HH, m0, k0, lane);
        if (k0 + 32 < HH) {
            __builtin_prefetch((const void*)(A + (long)(m0 + li) * HH + k0 + 32), 0, 0);
        }
#pragma unroll
        for (int j = 0; j < 4; ++j) {
            int nt = ng * 4 + j;
            if (nt >= VV / 16) break;          // uniform per wave: EXEC stays full
            v16bf b = load_tile16x32(BT, HH, nt << 4, k0, lane);
            c[j] = wmma_bf16(a, b, c[j]);
        }
    }
    const int ro = (lane >> 4) << 3;
#pragma unroll
    for (int j = 0; j < 4; ++j) {
        int nt = ng * 4 + j;
        if (nt >= VV / 16) break;
        int col = (nt << 4) + li;
        float bv = bias[col];
#pragma unroll
        for (int i = 0; i < 8; ++i) {
            int row = m0 + ro + i;
            out[(long)row * VV + col] = c[j][i] + bv;
        }
    }
}

// ---------------------------------------------------------------------------
// Host orchestration
// ---------------------------------------------------------------------------
extern "C" void kernel_launch(void* const* d_in, const int* in_sizes, int n_in,
                              void* d_out, int out_size, void* d_ws, size_t ws_size,
                              hipStream_t stream) {
    (void)in_sizes; (void)n_in; (void)out_size; (void)ws_size;

    const int*   tokens = (const int*)  d_in[0];
    const float* cnn    = (const float*)d_in[1];
    const float* emb    = (const float*)d_in[2];
    const float* w_in   = (const float*)d_in[3];
    const float* b_in   = (const float*)d_in[4];
    const float* w_r    = (const float*)d_in[5];
    const float* b_r    = (const float*)d_in[6];
    const float* w_u    = (const float*)d_in[7];
    const float* b_u    = (const float*)d_in[8];
    const float* w_h    = (const float*)d_in[9];
    const float* b_h    = (const float*)d_in[10];
    const float* w_out  = (const float*)d_in[11];
    const float* b_out  = (const float*)d_in[12];
    float* out = (float*)d_out;

    // --- workspace carve-up (all offsets deterministic) ---
    char* p = (char*)d_ws;
    auto carve = [&](size_t bytes) -> void* {
        void* r = (void*)p;
        p += (bytes + 255) & ~(size_t)255;
        return r;
    };
    bf16_t* w_inT  = (bf16_t*)carve((size_t)HH * FF * 2);        // [512,2048]
    bf16_t* wrT    = (bf16_t*)carve((size_t)LL * HH * DD * 2);   // [2][512,1024]
    bf16_t* wuT    = (bf16_t*)carve((size_t)LL * HH * DD * 2);
    bf16_t* whT    = (bf16_t*)carve((size_t)LL * HH * DD * 2);
    bf16_t* w_outT = (bf16_t*)carve((size_t)VV * HH * 2);        // [10000,512]
    bf16_t* cnn_bf = (bf16_t*)carve((size_t)BB * FF * 2);
    bf16_t* seqE   = (bf16_t*)carve((size_t)BB * SS * HH * 2);   // embeddings
    bf16_t* seqA   = (bf16_t*)carve((size_t)BB * SS * HH * 2);   // layer0 out
    bf16_t* seqB   = (bf16_t*)carve((size_t)BB * SS * HH * 2);   // layer1 out
    float*  h0f    = (float*) carve((size_t)BB * HH * 4);
    bf16_t* h0b    = (bf16_t*)carve((size_t)BB * HH * 2);
    float*  hf     = (float*) carve((size_t)BB * HH * 4);
    bf16_t* hbf    = (bf16_t*)carve((size_t)BB * HH * 2);
    float*  uf     = (float*) carve((size_t)BB * HH * 4);
    bf16_t* rh     = (bf16_t*)carve((size_t)BB * HH * 2);

    const dim3 tb(32, 8);

    // --- one-time-per-call weight conversion / transposition to bf16 [N,K] ---
    k_transpose_f32_bf16<<<dim3(HH / 32, FF / 32), tb, 0, stream>>>(w_in, w_inT, FF, HH);
    for (int l = 0; l < LL; ++l) {
        const size_t wOff = (size_t)l * DD * HH;    // src [D,H]
        const size_t tOff = (size_t)l * HH * DD;    // dst [H,D]
        k_transpose_f32_bf16<<<dim3(HH / 32, DD / 32), tb, 0, stream>>>(w_r + wOff, wrT + tOff, DD, HH);
        k_transpose_f32_bf16<<<dim3(HH / 32, DD / 32), tb, 0, stream>>>(w_u + wOff, wuT + tOff, DD, HH);
        k_transpose_f32_bf16<<<dim3(HH / 32, DD / 32), tb, 0, stream>>>(w_h + wOff, whT + tOff, DD, HH);
    }
    k_transpose_f32_bf16<<<dim3((VV + 31) / 32, HH / 32), tb, 0, stream>>>(w_out, w_outT, HH, VV);
    k_f32_to_bf16<<<(BB * FF + 255) / 256, 256, 0, stream>>>(cnn, cnn_bf, BB * FF);

    // --- embedding gather ---
    k_embed_gather<<<BB * SS, 256, 0, stream>>>(tokens, emb, seqE);

    // --- initial hidden state h0 ---
    k_wmma_h0<<<16, 256, 0, stream>>>(cnn_bf, w_inT, b_in, h0f, h0b);

    // --- GRU layers ---
    const bf16_t* seq_in = seqE;
    bf16_t*       seq_out = seqA;
    const long seqStride = (long)SS * HH;
    for (int l = 0; l < LL; ++l) {
        hipMemcpyAsync(hf,  h0f, (size_t)BB * HH * 4, hipMemcpyDeviceToDevice, stream);
        hipMemcpyAsync(hbf, h0b, (size_t)BB * HH * 2, hipMemcpyDeviceToDevice, stream);
        const bf16_t* wrTl = wrT + (size_t)l * HH * DD;
        const bf16_t* wuTl = wuT + (size_t)l * HH * DD;
        const bf16_t* whTl = whT + (size_t)l * HH * DD;
        const float*  brl  = b_r + (size_t)l * HH;
        const float*  bul  = b_u + (size_t)l * HH;
        const float*  bhl  = b_h + (size_t)l * HH;
        for (int t = 0; t < SS; ++t) {
            const bf16_t* xt = seq_in + (size_t)t * HH;
            k_gru_gates_ru<<<32, 256, 0, stream>>>(xt, seqStride, hbf,
                                                   wrTl, wuTl, brl, bul,
                                                   hf, uf, rh);
            k_gru_htilde_update<<<16, 256, 0, stream>>>(xt, seqStride, rh,
                                                        whTl, bhl, uf, hf, hbf,
                                                        seq_out + (size_t)t * HH,
                                                        seqStride);
        }
        seq_in = seq_out;
        seq_out = seqB;
    }

    // --- logits GEMM: [2048,512] x [512,10000] + bias ---
    k_wmma_logits<<<(128 * 157 + 7) / 8, 256, 0, stream>>>(seq_in, w_outT, b_out, out);
}
